// PVCNN2Base_4148938408196
// MI455X (gfx1250) — matrix operand order repaired and checked by reading, not compiled, not run on hardware
//
#include <hip/hip_runtime.h>
#include <hip/hip_bf16.h>

typedef __attribute__((ext_vector_type(16))) _Float16 v16h;
typedef __attribute__((ext_vector_type(8)))  _Float16 v8h;
typedef __attribute__((ext_vector_type(8)))  float    v8f;

#define B_ 8
#define C_ 64
#define N_ 16384
#define RR 32
#define R3 32768
#define GRP 8

// ---------------------------------------------------------------------------
// Weight pre-swizzle: w[o][i][t] f32  ->  Wg[t][o][i] f16   (t=0..26)
// ---------------------------------------------------------------------------
__global__ void k_prep_wconv(const float* __restrict__ w, _Float16* __restrict__ wg) {
    int i = blockIdx.x * 256 + threadIdx.x;         // 27*64*64 = 110592
    if (i >= 27 * 64 * 64) return;
    int kk = i & 63;
    int o  = (i >> 6) & 63;
    int t  = i >> 12;
    wg[i] = (_Float16)w[((size_t)(o * 64 + kk)) * 27 + t];
}

__global__ void k_prep_wp(const float* __restrict__ w, _Float16* __restrict__ wg) {
    int i = blockIdx.x * 256 + threadIdx.x;         // 2*64*32 = 4096
    if (i >= 2 * 64 * 32) return;
    int kk  = i & 31;
    int o   = (i >> 5) & 63;
    int c32 = i >> 11;
    wg[i] = (_Float16)w[o * 64 + c32 * 32 + kk];
}

__global__ void k_f32_to_f16(const float* __restrict__ x, _Float16* __restrict__ y, size_t n) {
    size_t i = (size_t)blockIdx.x * 256 + threadIdx.x;
    if (i < n) y[i] = (_Float16)x[i];
}

// ---------------------------------------------------------------------------
// Tiled transpose: X f32 [b][c=64][p in S]  ->  Y f16 [b][p][c=64]
// 64x64 tiles via LDS, coalesced on both sides.
// ---------------------------------------------------------------------------
__global__ __launch_bounds__(256) void k_transpose_f16(const float* __restrict__ X,
                                                       _Float16* __restrict__ Y, int S) {
    __shared__ _Float16 T[64][72];
    int tiles = S >> 6;
    int b  = blockIdx.x / tiles;
    int p0 = (blockIdx.x % tiles) << 6;
    int tid = threadIdx.x;
    {   // read: lanes sweep p (coalesced f32), c in chunks of 16 per thread-quad
        int p  = tid & 63;
        int c0 = (tid >> 6) * 16;
        const float* src = X + (size_t)b * 64 * S + p0;
        for (int u = 0; u < 16; ++u)
            T[p][c0 + u] = (_Float16)src[(size_t)(c0 + u) * S + p];
    }
    __syncthreads();
    {   // write: output tile is one contiguous 8KB block -> fully coalesced b128s
        _Float16* dst = Y + ((size_t)b * S + p0) * 64;
        int pl = tid >> 2, cc = (tid & 3) * 16;
        *(v8h*)(dst + pl * 64 + cc)     = *(const v8h*)&T[pl][cc];
        *(v8h*)(dst + pl * 64 + cc + 8) = *(const v8h*)&T[pl][cc + 8];
    }
}

// ---------------------------------------------------------------------------
// coords mean per (b,axis), then per-b max norm
// ---------------------------------------------------------------------------
__global__ void k_mean(const float* __restrict__ coords, float* __restrict__ meanb) {
    __shared__ float sh[256];
    int row = blockIdx.x;                            // b*3+axis
    const float* p = coords + (size_t)row * N_;
    float s = 0.f;
    for (int n = threadIdx.x; n < N_; n += 256) s += p[n];
    sh[threadIdx.x] = s; __syncthreads();
    for (int o = 128; o; o >>= 1) { if (threadIdx.x < o) sh[threadIdx.x] += sh[threadIdx.x + o]; __syncthreads(); }
    if (threadIdx.x == 0) meanb[row] = sh[0] / (float)N_;
}

__global__ void k_rmax(const float* __restrict__ coords, const float* __restrict__ meanb,
                       float* __restrict__ rmax) {
    __shared__ float sh[256];
    int b = blockIdx.x;
    float mx = meanb[b * 3 + 0], my = meanb[b * 3 + 1], mz = meanb[b * 3 + 2];
    const float* px = coords + (size_t)(b * 3 + 0) * N_;
    const float* py = coords + (size_t)(b * 3 + 1) * N_;
    const float* pz = coords + (size_t)(b * 3 + 2) * N_;
    float m = 0.f;
    for (int n = threadIdx.x; n < N_; n += 256) {
        float dx = px[n] - mx, dy = py[n] - my, dz = pz[n] - mz;
        m = fmaxf(m, sqrtf(dx * dx + dy * dy + dz * dz));
    }
    sh[threadIdx.x] = m; __syncthreads();
    for (int o = 128; o; o >>= 1) { if (threadIdx.x < o) sh[threadIdx.x] = fmaxf(sh[threadIdx.x], sh[threadIdx.x + o]); __syncthreads(); }
    if (threadIdx.x == 0) rmax[b] = sh[0];
}

__global__ void k_nc(const float* __restrict__ coords, const float* __restrict__ meanb,
                     const float* __restrict__ rmax, float* __restrict__ nc, int* __restrict__ idx) {
    int i = blockIdx.x * 256 + threadIdx.x;          // B*N
    if (i >= B_ * N_) return;
    int b = i / N_, n = i - b * N_;
    float inv = 1.f / (rmax[b] * 2.f);
    int vox[3];
    for (int a = 0; a < 3; ++a) {
        float v = (coords[(size_t)(b * 3 + a) * N_ + n] - meanb[b * 3 + a]) * inv + 0.5f;
        v *= (float)RR;
        v = fminf(fmaxf(v, 0.f), (float)(RR - 1));
        nc[(size_t)(b * 3 + a) * N_ + n] = v;
        vox[a] = (int)rintf(v);                       // round-half-even == jnp.round
    }
    idx[i] = (vox[0] * RR + vox[1]) * RR + vox[2];
}

// ---------------------------------------------------------------------------
// scatter-mean into voxel grid
// ---------------------------------------------------------------------------
__global__ void k_scatter(const float* __restrict__ feat, const int* __restrict__ idx,
                          float* __restrict__ gsum, float* __restrict__ cnt) {
    int i = blockIdx.x * 256 + threadIdx.x;          // B*N
    if (i >= B_ * N_) return;
    int b = i / N_, n = i - b * N_;
    int v = idx[i];
    atomicAdd(cnt + (size_t)b * R3 + v, 1.f);
    const float* fb = feat + (size_t)b * C_ * N_;
    float* gb = gsum + (size_t)b * C_ * R3;
    for (int c = 0; c < C_; ++c)
        atomicAdd(gb + (size_t)c * R3 + v, fb[(size_t)c * N_ + n]);
}

__global__ void k_finalize_vox(float* __restrict__ g, const float* __restrict__ cnt) {
    size_t i = (size_t)blockIdx.x * 256 + threadIdx.x;   // B*C*R3
    if (i >= (size_t)B_ * C_ * R3) return;
    int p = (int)(i & (R3 - 1));
    int b = (int)(i >> 21);                               // 64*32768 = 2^21
    g[i] = g[i] / fmaxf(cnt[(size_t)b * R3 + p], 1.f);
}

// ---------------------------------------------------------------------------
// 3x3x3 conv as implicit GEMM on v_wmma_f32_16x16x32_f16.
// Input Xt is voxel-major f16 [b][vox][ci]; the block stages a zero-padded
// 3x6x34 voxel halo (x64 ci) in LDS once, then the 27-tap K-loop feeds WMMA
// B-fragments directly from LDS (two ds_load_b128 per fragment, no branches).
// Block tile: 64(Cout) x 128(voxels), 8 waves.
// ---------------------------------------------------------------------------
#define HV (3 * 6 * 34)          // 612 halo voxels
__global__ __launch_bounds__(256) void k_conv3x3(const _Float16* __restrict__ Xt,
                                                 const _Float16* __restrict__ Wg,
                                                 const float* __restrict__ bias,
                                                 float* __restrict__ Y) {
    __shared__ _Float16 Xs[HV][72];    // [halo voxel][ci]  (pad 64->72 vs banks)
    __shared__ _Float16 Al[64][72];    // [co][ci] for current tap

    int b    = blockIdx.x >> 8;
    int tile = blockIdx.x & 255;
    int p0   = tile * 128;             // 4 full w-rows of one d-slice
    int d0   = p0 >> 10;
    int h0   = (p0 >> 5) & 31;

    int tid  = threadIdx.x;
    int lane = tid & 31;
    int wid  = tid >> 5;
    int mi   = wid & 3;
    int ni   = wid >> 2;
    int lr   = lane & 15;
    int hsel = lane >> 4;

    // ---- stage halo (coalesced 128B per voxel) ----
    for (int v = tid; v < HV; v += 256) {
        int dl = v / (6 * 34);
        int r  = v - dl * (6 * 34);
        int hl = r / 34;
        int wl = r - hl * 34;
        int dd = d0 + dl - 1, hh = h0 + hl - 1, ww = wl - 1;
        if ((unsigned)dd < 32u && (unsigned)hh < 32u && (unsigned)ww < 32u) {
            const _Float16* src = Xt + ((size_t)b * R3 + dd * 1024 + hh * 32 + ww) * 64;
            for (int u = 0; u < 8; ++u) *(v8h*)&Xs[v][u * 8] = *(const v8h*)(src + u * 8);
        } else {
            v8h z = {};
            for (int u = 0; u < 8; ++u) *(v8h*)&Xs[v][u * 8] = z;
        }
    }

    v8f acc0 = {}, acc1 = {}, acc2 = {}, acc3 = {};

    for (int t = 0; t < 27; ++t) {
        int kd = t / 9, kh = (t / 3) % 3, kw = t % 3;      // 0..2 (halo-local)
        __syncthreads();
        {   // A tile for this tap: 64x64 halves, 2 b128 per thread
            int m  = tid >> 2;
            int k0 = (tid & 3) * 16;
            const _Float16* src = Wg + ((size_t)t * 64 + m) * 64 + k0;
            *(v8h*)&Al[m][k0]     = *(const v8h*)src;
            *(v8h*)&Al[m][k0 + 8] = *(const v8h*)(src + 8);
        }
        __syncthreads();
        union { v16h v; v8h h[2]; } af, bf;
        for (int c32 = 0; c32 < 2; ++c32) {
            int ka = c32 * 32 + hsel * 16;                 // A: K {ka..ka+15}
            af.h[0] = *(const v8h*)&Al[mi * 16 + lr][ka];
            af.h[1] = *(const v8h*)&Al[mi * 16 + lr][ka + 8];
            int kb = c32 * 32 + hsel * 16;                 // B: K {kb..kb+15}
            int nb = ni * 64 + lr;
            int vrow = (kd * 6 + (nb >> 5) + kh) * 34 + (nb & 31) + kw;
            // j = 0..3 -> n offset +0,+16,+32,+48 : voxel rows shift by +16 w
            const _Float16* x0 = &Xs[vrow][kb];
            bf.h[0] = *(const v8h*)x0;                   bf.h[1] = *(const v8h*)(x0 + 8);
            acc0 = __builtin_amdgcn_wmma_f32_16x16x32_f16(false, af.v, false, bf.v, (short)0, acc0, false, false);
            const _Float16* x1 = &Xs[vrow + 16][kb];       // +16 voxels, same h row span
            bf.h[0] = *(const v8h*)x1;                   bf.h[1] = *(const v8h*)(x1 + 8);
            acc1 = __builtin_amdgcn_wmma_f32_16x16x32_f16(false, af.v, false, bf.v, (short)0, acc1, false, false);
            const _Float16* x2 = &Xs[vrow + 34][kb];       // +32 voxels = next h row
            bf.h[0] = *(const v8h*)x2;                   bf.h[1] = *(const v8h*)(x2 + 8);
            acc2 = __builtin_amdgcn_wmma_f32_16x16x32_f16(false, af.v, false, bf.v, (short)0, acc2, false, false);
            const _Float16* x3 = &Xs[vrow + 50][kb];       // +48 voxels
            bf.h[0] = *(const v8h*)x3;                   bf.h[1] = *(const v8h*)(x3 + 8);
            acc3 = __builtin_amdgcn_wmma_f32_16x16x32_f16(false, af.v, false, bf.v, (short)0, acc3, false, false);
        }
    }
    // C layout: VGPR r -> M = mi*16 + r + 8*hsel, N = lr
    v8f* accs[4] = { &acc0, &acc1, &acc2, &acc3 };
    for (int j = 0; j < 4; ++j) {
        int n = p0 + ni * 64 + j * 16 + lr;
        for (int r = 0; r < 8; ++r) {
            int m = mi * 16 + r + hsel * 8;
            Y[(size_t)(b * C_ + m) * R3 + n] = (*accs[j])[r] + bias[m];
        }
    }
}

// NOTE on vrow arithmetic above: voxel n = ni*64 + j*16 + lr spans w 0..31 within
// 4 h-rows; n and n+16 are either the same h-row (w+16) or the next one. Since
// Xs rows are (h,w) linear with stride 34 per h-row and +1 per w, halo voxel
// linear offset for +16 voxels is +16 (same row) or +34-16+... -- both cases
// collapse to: voxel (n>>5) row, (n&31) col. For j increments of 16:
//   j*16 added to n changes (n>>5)*34 + (n&31) by exactly +16 (if w<16) or
//   +34-32+16+... : (n+16)>>5*34 + ((n+16)&31) - [(n>>5)*34 + (n&31)]
//   = 16 (w<16) or 18 (w>=16). lr<16 always, and ni*64 is a multiple of 32,
//   so (nb&31)=lr<16: +16 -> +16 ; +32 -> +34 ; +48 -> +50.  (as coded)

// ---------------------------------------------------------------------------
// point MLP: Y[b] = Wp(64x64) @ X[b](64xN) + bias
// ---------------------------------------------------------------------------
__global__ __launch_bounds__(256) void k_pgemm(const _Float16* __restrict__ Xh,
                                               const _Float16* __restrict__ Wg,
                                               const float* __restrict__ bias,
                                               float* __restrict__ Y) {
    __shared__ _Float16 Al[64][40];
    __shared__ _Float16 Bl[128][40];
    int blk  = blockIdx.x;
    int b    = blk >> 7;
    int p0   = (blk & 127) * 128;
    int tid  = threadIdx.x;
    int lane = tid & 31;
    int wid  = tid >> 5;
    int mi   = wid & 3;
    int ni   = wid >> 2;
    int lr   = lane & 15;
    int hsel = lane >> 4;

    v8f acc0 = {}, acc1 = {}, acc2 = {}, acc3 = {};
    for (int c32 = 0; c32 < 2; ++c32) {
        __syncthreads();
        {
            int m  = tid >> 2;
            int ko = (tid & 3) * 8;
            *(v8h*)&Al[m][ko] = *(const v8h*)(Wg + ((size_t)(c32 * 64 + m) * 32 + ko));
        }
        {
            int kk = tid >> 3;
            int g  = tid & 7;
            const _Float16* xc = Xh + (size_t)(b * C_ + c32 * 32 + kk) * N_ + p0;
            for (int u = 0; u < 16; ++u) Bl[g * 16 + u][kk] = xc[g * 16 + u];
        }
        __syncthreads();
        union { v16h v; v8h h[2]; } af, bf;
        int ka = hsel * 16;
        af.h[0] = *(const v8h*)&Al[mi * 16 + lr][ka];
        af.h[1] = *(const v8h*)&Al[mi * 16 + lr][ka + 8];
        int kb = hsel * 16;
        int nb = ni * 64 + lr;
        bf.h[0] = *(const v8h*)&Bl[nb][kb];      bf.h[1] = *(const v8h*)&Bl[nb][kb + 8];
        acc0 = __builtin_amdgcn_wmma_f32_16x16x32_f16(false, af.v, false, bf.v, (short)0, acc0, false, false);
        bf.h[0] = *(const v8h*)&Bl[nb + 16][kb]; bf.h[1] = *(const v8h*)&Bl[nb + 16][kb + 8];
        acc1 = __builtin_amdgcn_wmma_f32_16x16x32_f16(false, af.v, false, bf.v, (short)0, acc1, false, false);
        bf.h[0] = *(const v8h*)&Bl[nb + 32][kb]; bf.h[1] = *(const v8h*)&Bl[nb + 32][kb + 8];
        acc2 = __builtin_amdgcn_wmma_f32_16x16x32_f16(false, af.v, false, bf.v, (short)0, acc2, false, false);
        bf.h[0] = *(const v8h*)&Bl[nb + 48][kb]; bf.h[1] = *(const v8h*)&Bl[nb + 48][kb + 8];
        acc3 = __builtin_amdgcn_wmma_f32_16x16x32_f16(false, af.v, false, bf.v, (short)0, acc3, false, false);
    }
    v8f* accs[4] = { &acc0, &acc1, &acc2, &acc3 };
    for (int j = 0; j < 4; ++j) {
        int n = p0 + ni * 64 + j * 16 + lr;
        for (int r = 0; r < 8; ++r) {
            int m = mi * 16 + r + hsel * 8;
            Y[(size_t)(b * C_ + m) * N_ + n] = (*accs[j])[r] + bias[m];
        }
    }
}

// ---------------------------------------------------------------------------
// GroupNorm stats (mean, invstd) per (b, group)
// ---------------------------------------------------------------------------
__global__ void k_gn_stats(const float* __restrict__ X, float* __restrict__ stats, int S) {
    __shared__ float sh[512];
    int b = blockIdx.x >> 3, g = blockIdx.x & 7;
    const float* base = X + (size_t)(b * C_ + g * (C_ / GRP)) * S;
    int tot = (C_ / GRP) * S;
    float s = 0.f, ss = 0.f;
    for (int i = threadIdx.x; i < tot; i += 256) { float v = base[i]; s += v; ss += v * v; }
    sh[threadIdx.x] = s; sh[256 + threadIdx.x] = ss; __syncthreads();
    for (int o = 128; o; o >>= 1) {
        if (threadIdx.x < o) { sh[threadIdx.x] += sh[threadIdx.x + o]; sh[256 + threadIdx.x] += sh[256 + threadIdx.x + o]; }
        __syncthreads();
    }
    if (threadIdx.x == 0) {
        float mu  = sh[0] / (float)tot;
        float var = sh[256] / (float)tot - mu * mu;
        stats[blockIdx.x * 2]     = mu;
        stats[blockIdx.x * 2 + 1] = rsqrtf(var + 1e-5f);
    }
}

__global__ void k_gn_swish(float* __restrict__ X, const float* __restrict__ stats,
                           const float* __restrict__ gamma, const float* __restrict__ beta, int S) {
    size_t i = (size_t)blockIdx.x * 256 + threadIdx.x;
    size_t tot = (size_t)B_ * C_ * S;
    if (i >= tot) return;
    int c = (int)((i / S) & (C_ - 1));
    int b = (int)(i / ((size_t)C_ * S));
    int g = c >> 3;
    float mu = stats[(b * GRP + g) * 2], is = stats[(b * GRP + g) * 2 + 1];
    float v = (X[i] - mu) * is * gamma[c] + beta[c];
    X[i] = v / (1.f + __expf(-v));         // swish
}

// ---------------------------------------------------------------------------
// trilinear devoxelize + add point branch
// ---------------------------------------------------------------------------
__global__ void k_devox_add(const float* __restrict__ H, const float* __restrict__ P,
                            const float* __restrict__ nc, float* __restrict__ out) {
    int i = blockIdx.x * 256 + threadIdx.x;   // B*N
    if (i >= B_ * N_) return;
    int b = i / N_, n = i - b * N_;
    float nx = nc[(size_t)(b * 3 + 0) * N_ + n];
    float ny = nc[(size_t)(b * 3 + 1) * N_ + n];
    float nz = nc[(size_t)(b * 3 + 2) * N_ + n];
    int x0 = (int)floorf(nx), y0 = (int)floorf(ny), z0 = (int)floorf(nz);
    int x1 = min(x0 + 1, RR - 1), y1 = min(y0 + 1, RR - 1), z1 = min(z0 + 1, RR - 1);
    float fx = nx - floorf(nx), fy = ny - floorf(ny), fz = nz - floorf(nz);
    float wx0 = 1.f - fx, wy0 = 1.f - fy, wz0 = 1.f - fz;
    int   id[8];
    float w[8];
    id[0] = (x0 * RR + y0) * RR + z0; w[0] = wx0 * wy0 * wz0;
    id[1] = (x0 * RR + y0) * RR + z1; w[1] = wx0 * wy0 * fz;
    id[2] = (x0 * RR + y1) * RR + z0; w[2] = wx0 * fy  * wz0;
    id[3] = (x0 * RR + y1) * RR + z1; w[3] = wx0 * fy  * fz;
    id[4] = (x1 * RR + y0) * RR + z0; w[4] = fx  * wy0 * wz0;
    id[5] = (x1 * RR + y0) * RR + z1; w[5] = fx  * wy0 * fz;
    id[6] = (x1 * RR + y1) * RR + z0; w[6] = fx  * fy  * wz0;
    id[7] = (x1 * RR + y1) * RR + z1; w[7] = fx  * fy  * fz;
    const float* hb = H + (size_t)b * C_ * R3;
    const float* pb = P + (size_t)b * C_ * N_;
    float* ob = out + (size_t)b * C_ * N_;
    for (int c = 0; c < C_; ++c) {
        const float* hc = hb + (size_t)c * R3;
        float acc = pb[(size_t)c * N_ + n];
        for (int k = 0; k < 8; ++k) acc += w[k] * hc[id[k]];
        ob[(size_t)c * N_ + n] = acc;
    }
}

// ---------------------------------------------------------------------------
// host side
// ---------------------------------------------------------------------------
extern "C" void kernel_launch(void* const* d_in, const int* in_sizes, int n_in,
                              void* d_out, int out_size, void* d_ws, size_t ws_size,
                              hipStream_t stream) {
    const float* feat   = (const float*)d_in[0];
    const float* coords = (const float*)d_in[1];
    const float* w1  = (const float*)d_in[2];
    const float* b1  = (const float*)d_in[3];
    const float* g1  = (const float*)d_in[4];
    const float* be1 = (const float*)d_in[5];
    const float* w2  = (const float*)d_in[6];
    const float* b2  = (const float*)d_in[7];
    const float* g2  = (const float*)d_in[8];
    const float* be2 = (const float*)d_in[9];
    const float* wp  = (const float*)d_in[10];
    const float* bp  = (const float*)d_in[11];
    const float* gp  = (const float*)d_in[12];
    const float* bep = (const float*)d_in[13];
    float* out = (float*)d_out;

    char* ws = (char*)d_ws;
    constexpr size_t SZ_GRIDF = (size_t)B_ * C_ * R3 * 4;   // 64 MB
    constexpr size_t SZ_GRIDH = SZ_GRIDF / 2;               // 32 MB
    constexpr size_t SZ_FEATH = (size_t)B_ * C_ * N_ * 2;   // 16 MB
    constexpr size_t SZ_POINT = (size_t)B_ * C_ * N_ * 4;   // 32 MB
    constexpr size_t SZ_CNT   = (size_t)B_ * R3 * 4;
    constexpr size_t SZ_NC    = (size_t)B_ * 3 * N_ * 4;
    constexpr size_t SZ_IDX   = (size_t)B_ * N_ * 4;
    constexpr size_t SZ_WCONV = (size_t)27 * 64 * 64 * 2;
    constexpr size_t SZ_WP    = (size_t)2 * 64 * 32 * 2;

    size_t o = 0;
    auto take = [&](size_t bytes) { size_t r = o; o += (bytes + 255) & ~(size_t)255; return r; };
    float*    gA    = (float*)   (ws + take(SZ_GRIDF));   // voxel sums -> grid -> conv2 out (h2)
    float*    gB    = (float*)   (ws + take(SZ_GRIDF));   // conv1 out (h1)
    _Float16* gH    = (_Float16*)(ws + take(SZ_GRIDH));   // voxel-major f16 conv input (reused)
    _Float16* featH = (_Float16*)(ws + take(SZ_FEATH));
    float*    pointP= (float*)   (ws + take(SZ_POINT));
    float*    cnt   = (float*)   (ws + take(SZ_CNT));
    float*    ncb   = (float*)   (ws + take(SZ_NC));
    int*      idxb  = (int*)     (ws + take(SZ_IDX));
    _Float16* w1h   = (_Float16*)(ws + take(SZ_WCONV));
    _Float16* w2h   = (_Float16*)(ws + take(SZ_WCONV));
    _Float16* wph   = (_Float16*)(ws + take(SZ_WP));
    float*    meanb = (float*)   (ws + take(256));
    float*    rmaxb = (float*)   (ws + take(256));
    float*    st1   = (float*)   (ws + take(512));
    float*    st2   = (float*)   (ws + take(512));
    float*    stP   = (float*)   (ws + take(512));
    (void)ws_size; (void)n_in; (void)in_sizes; (void)out_size;

    hipMemsetAsync(gA, 0, SZ_GRIDF, stream);
    hipMemsetAsync(cnt, 0, SZ_CNT, stream);

    // precision / layout prep
    k_prep_wconv<<<(27 * 64 * 64 + 255) / 256, 256, 0, stream>>>(w1, w1h);
    k_prep_wconv<<<(27 * 64 * 64 + 255) / 256, 256, 0, stream>>>(w2, w2h);
    k_prep_wp<<<(2 * 64 * 32 + 255) / 256, 256, 0, stream>>>(wp, wph);
    k_f32_to_f16<<<(unsigned)(((size_t)B_ * C_ * N_ + 255) / 256), 256, 0, stream>>>(feat, featH, (size_t)B_ * C_ * N_);

    // voxelize
    k_mean<<<B_ * 3, 256, 0, stream>>>(coords, meanb);
    k_rmax<<<B_, 256, 0, stream>>>(coords, meanb, rmaxb);
    k_nc<<<(B_ * N_ + 255) / 256, 256, 0, stream>>>(coords, meanb, rmaxb, ncb, idxb);
    k_scatter<<<(B_ * N_ + 255) / 256, 256, 0, stream>>>(feat, idxb, gA, cnt);
    k_finalize_vox<<<(unsigned)(((size_t)B_ * C_ * R3 + 255) / 256), 256, 0, stream>>>(gA, cnt);
    k_transpose_f16<<<B_ * (R3 / 64), 256, 0, stream>>>(gA, gH, R3);

    // conv1 -> GN1 -> swish -> transpose to voxel-major f16 for conv2
    k_conv3x3<<<B_ * 256, 256, 0, stream>>>(gH, w1h, b1, gB);
    k_gn_stats<<<B_ * GRP, 256, 0, stream>>>(gB, st1, R3);
    k_gn_swish<<<(unsigned)(((size_t)B_ * C_ * R3 + 255) / 256), 256, 0, stream>>>(gB, st1, g1, be1, R3);
    k_transpose_f16<<<B_ * (R3 / 64), 256, 0, stream>>>(gB, gH, R3);

    // conv2 -> GN2 -> swish (overwrites gA; devox reads f32)
    k_conv3x3<<<B_ * 256, 256, 0, stream>>>(gH, w2h, b2, gA);
    k_gn_stats<<<B_ * GRP, 256, 0, stream>>>(gA, st2, R3);
    k_gn_swish<<<(unsigned)(((size_t)B_ * C_ * R3 + 255) / 256), 256, 0, stream>>>(gA, st2, g2, be2, R3);

    // point branch: GEMM -> GN -> swish
    k_pgemm<<<B_ * (N_ / 128), 256, 0, stream>>>(featH, wph, bp, pointP);
    k_gn_stats<<<B_ * GRP, 256, 0, stream>>>(pointP, stP, N_);
    k_gn_swish<<<(unsigned)(((size_t)B_ * C_ * N_ + 255) / 256), 256, 0, stream>>>(pointP, stP, gp, bep, N_);

    // devoxelize + add
    k_devox_add<<<(B_ * N_ + 255) / 256, 256, 0, stream>>>(gA, pointP, ncb, out);
}
